// WanSelfAttention_11132555231260
// MI455X (gfx1250) — compile-verified, matrix operand order checked by reading, not run on hardware
//
#include <hip/hip_runtime.h>
#include <stdint.h>

// ---------------- problem constants ----------------
#define DIM    1536
#define HEADS  12
#define HD     128            // head dim
#define HALF   64             // HD/2 rope pairs
#define NFRM   4
#define NH     30
#define NW     52
#define SEQ    (NFRM*NH*NW)   // 6240
#define NCHUNK 16
#define CSZ    (SEQ/NCHUNK)   // 390
#define KEEP   624            // S - int(0.9*S)
#define CF     22
#define CHD    21
#define CWD    21
#define EPSN   1e-6f
#define JT     6              // key-tiles per wave in attn_logits (390/16 = 65*6)

typedef __attribute__((ext_vector_type(16))) _Float16 v16h;
typedef __attribute__((ext_vector_type(8)))  _Float16 v8h;
typedef __attribute__((ext_vector_type(8)))  float    v8f;

// CDNA5 16-bit fragment: lane's 16 elements are K = {8kh..8kh+7} U {16+8kh..23+8kh}
// => two contiguous 16-byte chunks when K is the fastest-varying axis.
__device__ __forceinline__ v16h load_frag(const _Float16* base /*16B aligned, at k0*/, int kh) {
    v8h lo = *(const v8h*)(base + 8 * kh);
    v8h hi = *(const v8h*)(base + 16 + 8 * kh);
    return __builtin_shufflevector(lo, hi, 0, 1, 2, 3, 4, 5, 6, 7, 8, 9, 10, 11, 12, 13, 14, 15);
}

__device__ __forceinline__ v8f wmma16(v16h a, v16h b, v8f c) {
    return __builtin_amdgcn_wmma_f32_16x16x32_f16(false, a, false, b, (short)0, c, false, false);
}

__device__ __forceinline__ float bsum(float v, float* sh) {
    int t = threadIdx.x; sh[t] = v; __syncthreads();
    for (int s = 128; s > 0; s >>= 1) { if (t < s) sh[t] += sh[t + s]; __syncthreads(); }
    float r = sh[0]; __syncthreads(); return r;
}
__device__ __forceinline__ float bmaxr(float v, float* sh) {
    int t = threadIdx.x; sh[t] = v; __syncthreads();
    for (int s = 128; s > 0; s >>= 1) { if (t < s) sh[t] = fmaxf(sh[t], sh[t + s]); __syncthreads(); }
    float r = sh[0]; __syncthreads(); return r;
}
__device__ __forceinline__ unsigned fkey(float x) {
    unsigned u = __float_as_uint(x);
    return (u & 0x80000000u) ? ~u : (u | 0x80000000u);
}

// ---------------- f32 -> f16 convert ----------------
__global__ void cvt_f16(const float* __restrict__ in, _Float16* __restrict__ out, long n) {
    long i = (long)blockIdx.x * blockDim.x + threadIdx.x;
    long stride = (long)gridDim.x * blockDim.x;
    for (; i < n; i += stride) out[i] = (_Float16)in[i];
}

// ---------------- fused convert + transpose: out[c*R + r] = (f16) in[r*C + c] ----------------
// grid (C/32, R/32), block 256. LDS-tiled, padded against bank conflicts.
__global__ void transpose_cvt(const float* __restrict__ in, _Float16* __restrict__ out,
                              int R, int C) {
    __shared__ float tile[32][33];
    const int bx = blockIdx.x * 32;           // col base
    const int by = blockIdx.y * 32;           // row base
    const int tx = threadIdx.x & 31, ty = threadIdx.x >> 5;   // 32 x 8
#pragma unroll
    for (int i = 0; i < 32; i += 8)
        tile[ty + i][tx] = in[(size_t)(by + ty + i) * C + bx + tx];
    __syncthreads();
#pragma unroll
    for (int i = 0; i < 32; i += 8)
        out[(size_t)(bx + ty + i) * R + by + tx] = (_Float16)tile[tx][ty + i];
}

// ---------------- WMMA GEMM: C[M,N] = A[M,K](f16 row-major) @ Bt[N,K](f16, pre-transposed) + bias --
// block = 256 (8 waves); wave owns 32x64 tile: 2 A-frags x 4 B-frags -> 8 WMMA / 12 b128 loads.
__global__ void gemm_xw(const _Float16* __restrict__ A, const _Float16* __restrict__ Bt,
                        const float* __restrict__ bias, float* __restrict__ C,
                        int M, int N, int K) {
    const int lane = threadIdx.x & 31, wave = threadIdx.x >> 5;
    const int m0 = blockIdx.y * 256 + wave * 32;
    const int n0 = blockIdx.x * 64;
    const int kh = lane >> 4, l15 = lane & 15;
    const v16h zero16 = {};
    const int mA = m0 + l15, mB = m0 + 16 + l15;
    const int mcA = (mA < M) ? mA : (M - 1);
    const int mcB = (mB < M) ? mB : (M - 1);
    const bool vA = (mA < M), vB = (mB < M);
    const _Float16* arow0 = A + (size_t)mcA * K;
    const _Float16* arow1 = A + (size_t)mcB * K;
    v8f z = {};
    v8f acc[2][4] = { { z, z, z, z }, { z, z, z, z } };
    for (int k0 = 0; k0 < K; k0 += 32) {
        v16h a0 = load_frag(arow0 + k0, kh); a0 = vA ? a0 : zero16;
        v16h a1 = load_frag(arow1 + k0, kh); a1 = vB ? a1 : zero16;
#pragma unroll
        for (int nt = 0; nt < 4; ++nt) {
            const int n = n0 + nt * 16 + l15;
            v16h b = load_frag(Bt + (size_t)n * K + k0, kh);
            acc[0][nt] = wmma16(a0, b, acc[0][nt]);
            acc[1][nt] = wmma16(a1, b, acc[1][nt]);
        }
    }
#pragma unroll
    for (int half = 0; half < 2; ++half) {
#pragma unroll
        for (int nt = 0; nt < 4; ++nt) {
            const int n = n0 + nt * 16 + l15;
#pragma unroll
            for (int r = 0; r < 8; ++r) {
                int row = m0 + half * 16 + r + 8 * kh;
                if (row < M) C[(size_t)row * N + n] = acc[half][nt][r] + bias[n];
            }
        }
    }
}

// ---------------- RMS-norm(q,k) + factorized 3D RoPE --------------------------------------------
__global__ void norm_rope(const float* __restrict__ yq, const float* __restrict__ yk,
                          const float* __restrict__ gq, const float* __restrict__ gk,
                          const float* __restrict__ fcos, const float* __restrict__ fsin,
                          _Float16* __restrict__ rq, _Float16* __restrict__ rk) {
    __shared__ float sh[256];
    const int s = blockIdx.x, t = threadIdx.x;
    const float* q = yq + (size_t)s * DIM;
    const float* k = yk + (size_t)s * DIM;
    float sq = 0.f, sk = 0.f;
    for (int d = t; d < DIM; d += 256) { float a = q[d]; sq += a * a; float b = k[d]; sk += b * b; }
    sq = bsum(sq, sh); sk = bsum(sk, sh);
    const float rmq = rsqrtf(sq / (float)DIM + EPSN);
    const float rmk = rsqrtf(sk / (float)DIM + EPSN);
    const int f = s / (NH * NW), rem = s % (NH * NW), hh = rem / NW, ww = rem % NW;
    for (int p = t; p < DIM / 2; p += 256) {
        const int j = p & (HALF - 1);                       // rope pair index in head
        const int row = (j < CF) ? f : ((j < CF + CHD) ? hh : ww);
        const float c = fcos[row * HALF + j], sn = fsin[row * HALF + j];
        const int d0 = 2 * p;
        const float aq = q[d0] * rmq * gq[d0], bq = q[d0 + 1] * rmq * gq[d0 + 1];
        const float ak = k[d0] * rmk * gk[d0], bk = k[d0 + 1] * rmk * gk[d0 + 1];
        rq[(size_t)s * DIM + d0]     = (_Float16)(aq * c - bq * sn);
        rq[(size_t)s * DIM + d0 + 1] = (_Float16)(aq * sn + bq * c);
        rk[(size_t)s * DIM + d0]     = (_Float16)(ak * c - bk * sn);
        rk[(size_t)s * DIM + d0 + 1] = (_Float16)(ak * sn + bk * c);
    }
}

// ---------------- logits: wave owns 16 queries x 96 keys (A-frag reused x6) -----------------------
__global__ void attn_logits(const _Float16* __restrict__ rq, const _Float16* __restrict__ rk,
                            float* __restrict__ L, int chunk) {
    const int lane = threadIdx.x;
    const int j0 = blockIdx.x * (16 * JT), i0 = blockIdx.y * 16, h = blockIdx.z;
    const int kh = lane >> 4, l15 = lane & 15;
    const int m = i0 + l15;
    const int mc = (m < CSZ) ? m : (CSZ - 1);
    const bool mv = (m < CSZ);
    const v16h zero16 = {};
    v8f acc[JT] = {};
    const _Float16* qrow = rq + (size_t)(chunk * CSZ + mc) * DIM + h * HD;
#pragma unroll
    for (int k0 = 0; k0 < HD; k0 += 32) {
        v16h a = load_frag(qrow + k0, kh);
        a = mv ? a : zero16;
#pragma unroll
        for (int jt = 0; jt < JT; ++jt) {
            const _Float16* krow = rk + (size_t)(j0 + jt * 16 + l15) * DIM + h * HD;
            v16h b = load_frag(krow + k0, kh);
            acc[jt] = wmma16(a, b, acc[jt]);
        }
    }
    const float scale = 0.088388347648318447f; // 1/sqrt(128)
#pragma unroll
    for (int jt = 0; jt < JT; ++jt) {
        const int n = j0 + jt * 16 + l15;
#pragma unroll
        for (int r = 0; r < 8; ++r) {
            int row = i0 + r + 8 * kh;
            if (row < CSZ) L[((size_t)h * CSZ + row) * SEQ + n] = acc[jt][r] * scale;
        }
    }
}

// ---------------- exact top-k + masked softmax, one block per (h,i) row ---------------------------
// Row (25KB) staged into LDS ONCE via gfx1250 async load-to-LDS; 34 selection passes then hit LDS.
__global__ void topk_softmax(const float* __restrict__ L, _Float16* __restrict__ P) {
    __shared__ float lrow[SEQ];           // 24.96 KB
    __shared__ float sh[256];
    const int t = threadIdx.x;
    const float* l = L + (size_t)blockIdx.x * SEQ;

    // async stage: SEQ/4 = 1560 x b128 chunks
    for (int c = t; c < SEQ / 4; c += 256) {
        unsigned ldsoff = (unsigned)(uintptr_t)(&lrow[4 * c]);      // LDS aperture: addr[31:0]
        uint64_t gaddr  = (uint64_t)(uintptr_t)(l + 4 * c);
        asm volatile("global_load_async_to_lds_b128 %0, %1, off"
                     :: "v"(ldsoff), "v"(gaddr) : "memory");
    }
    asm volatile("s_wait_asynccnt 0" ::: "memory");
    __syncthreads();

    // row max
    float mx = -3.4e38f;
    for (int j = t; j < SEQ; j += 256) mx = fmaxf(mx, lrow[j]);
    mx = bmaxr(mx, sh);

    // largest key T with count(key >= T) >= KEEP  (== key of the KEEP-th largest value)
    unsigned ans = 0u;
    for (int b = 31; b >= 0; --b) {
        unsigned cand = ans | (1u << b);
        float cnt = 0.f;
        for (int j = t; j < SEQ; j += 256) cnt += (fkey(lrow[j]) >= cand) ? 1.f : 0.f;
        cnt = bsum(cnt, sh);
        if (cnt >= (float)KEEP) ans = cand;
    }

    float sum = 0.f;
    for (int j = t; j < SEQ; j += 256) {
        float x = lrow[j];
        if (fkey(x) >= ans) sum += __expf(x - mx);
    }
    sum = bsum(sum, sh);
    const float inv = 1.f / sum;

    _Float16* p = P + (size_t)blockIdx.x * SEQ;
    for (int j = t; j < SEQ; j += 256) {
        float x = lrow[j];
        p[j] = (fkey(x) >= ans) ? (_Float16)(__expf(x - mx) * inv) : (_Float16)0.f;
    }
}

// ---------------- PV: wave owns 16 queries x 128 (full head dim), A-frag reused x8 ----------------
__global__ void attn_pv(const _Float16* __restrict__ P, const _Float16* __restrict__ vt,
                        _Float16* __restrict__ ao, int chunk) {
    const int lane = threadIdx.x;
    const int i0 = blockIdx.x * 16, h = blockIdx.y;
    const int kh = lane >> 4, l15 = lane & 15;
    const int m = i0 + l15;
    const int mc = (m < CSZ) ? m : (CSZ - 1);
    const bool mv = (m < CSZ);
    const v16h zero16 = {};
    v8f acc[8] = {};
    const _Float16* prow = P + ((size_t)h * CSZ + mc) * SEQ;
    const _Float16* vbase = vt + (size_t)(h * HD + l15) * SEQ;
    for (int k0 = 0; k0 < SEQ; k0 += 32) {
        v16h a = load_frag(prow + k0, kh);
        a = mv ? a : zero16;
#pragma unroll
        for (int nt = 0; nt < 8; ++nt) {
            v16h b = load_frag(vbase + (size_t)(nt * 16) * SEQ + k0, kh);
            acc[nt] = wmma16(a, b, acc[nt]);
        }
    }
#pragma unroll
    for (int nt = 0; nt < 8; ++nt) {
#pragma unroll
        for (int r = 0; r < 8; ++r) {
            int row = i0 + r + 8 * kh;
            if (row < CSZ)
                ao[(size_t)(chunk * CSZ + row) * DIM + h * HD + nt * 16 + l15] = (_Float16)acc[nt][r];
        }
    }
}

// ---------------- host launcher ----------------
extern "C" void kernel_launch(void* const* d_in, const int* in_sizes, int n_in,
                              void* d_out, int out_size, void* d_ws, size_t ws_size,
                              hipStream_t stream) {
    const float* x    = (const float*)d_in[0];
    const float* fcos = (const float*)d_in[1];
    const float* fsin = (const float*)d_in[2];
    const float* Wq = (const float*)d_in[3];  const float* bq = (const float*)d_in[4];
    const float* Wk = (const float*)d_in[5];  const float* bk = (const float*)d_in[6];
    const float* Wv = (const float*)d_in[7];  const float* bv = (const float*)d_in[8];
    const float* Wo = (const float*)d_in[9];  const float* bo = (const float*)d_in[10];
    const float* gq = (const float*)d_in[11]; const float* gk = (const float*)d_in[12];
    float* out = (float*)d_out;

    char* w = (char*)d_ws;
    size_t off = 0;
    auto carve = [&](size_t bytes) { char* p = w + off; off = (off + bytes + 255) & ~(size_t)255; return p; };
    _Float16* xh  = (_Float16*)carve((size_t)SEQ * DIM * 2);
    _Float16* wqt = (_Float16*)carve((size_t)DIM * DIM * 2);   // Wq^T f16
    _Float16* wkt = (_Float16*)carve((size_t)DIM * DIM * 2);
    _Float16* wvt = (_Float16*)carve((size_t)DIM * DIM * 2);
    _Float16* wot = (_Float16*)carve((size_t)DIM * DIM * 2);
    float*    yq  = (float*)   carve((size_t)SEQ * DIM * 4);
    float*    yk  = (float*)   carve((size_t)SEQ * DIM * 4);
    float*    yv  = (float*)   carve((size_t)SEQ * DIM * 4);
    _Float16* rqh = (_Float16*)carve((size_t)SEQ * DIM * 2);
    _Float16* rkh = (_Float16*)carve((size_t)SEQ * DIM * 2);
    _Float16* vt  = (_Float16*)carve((size_t)SEQ * DIM * 2);   // V^T: [DIM][SEQ]
    float*    Lg  = (float*)   carve((size_t)HEADS * CSZ * SEQ * 4);
    _Float16* Pg  = (_Float16*)carve((size_t)HEADS * CSZ * SEQ * 2);
    _Float16* aoh = (_Float16*)carve((size_t)SEQ * DIM * 2);

    // 1) convert / pre-transpose operands so every WMMA fragment is 2x b128 per lane
    cvt_f16<<<4096, 256, 0, stream>>>(x, xh, (long)SEQ * DIM);
    dim3 wtg(DIM / 32, DIM / 32);
    transpose_cvt<<<wtg, 256, 0, stream>>>(Wq, wqt, DIM, DIM);
    transpose_cvt<<<wtg, 256, 0, stream>>>(Wk, wkt, DIM, DIM);
    transpose_cvt<<<wtg, 256, 0, stream>>>(Wv, wvt, DIM, DIM);
    transpose_cvt<<<wtg, 256, 0, stream>>>(Wo, wot, DIM, DIM);

    // 2) QKV projections (WMMA)
    dim3 ggrid(DIM / 64, (SEQ + 255) / 256);
    gemm_xw<<<ggrid, 256, 0, stream>>>(xh, wqt, bq, yq, SEQ, DIM, DIM);
    gemm_xw<<<ggrid, 256, 0, stream>>>(xh, wkt, bk, yk, SEQ, DIM, DIM);
    gemm_xw<<<ggrid, 256, 0, stream>>>(xh, wvt, bv, yv, SEQ, DIM, DIM);

    // 3) rms-norm + rope (q,k); transpose V for the PV matmul
    norm_rope<<<SEQ, 256, 0, stream>>>(yq, yk, gq, gk, fcos, fsin, rqh, rkh);
    dim3 vtg(DIM / 32, SEQ / 32);
    transpose_cvt<<<vtg, 256, 0, stream>>>(yv, vt, SEQ, DIM);

    // 4) chunked top-k attention
    dim3 lgrid(SEQ / (16 * JT), (CSZ + 15) / 16, HEADS);   // 65 x 25 x 12
    dim3 pvgrid((CSZ + 15) / 16, HEADS);                   // 25 x 12
    for (int c = 0; c < NCHUNK; ++c) {
        attn_logits <<<lgrid, 32, 0, stream>>>(rqh, rkh, Lg, c);
        topk_softmax<<<HEADS * CSZ, 256, 0, stream>>>(Lg, Pg);
        attn_pv     <<<pvgrid, 32, 0, stream>>>(Pg, vt, aoh, c);
    }

    // 5) output projection -> f32 d_out
    gemm_xw<<<ggrid, 256, 0, stream>>>(aoh, wot, bo, out, SEQ, DIM, DIM);
}
